// MemoryOptimizedBaselineModel_33440615367066
// MI455X (gfx1250) — matrix operand order repaired and checked by reading, not compiled, not run on hardware
//
#include <hip/hip_runtime.h>
#include <math.h>

#define DEV __device__ __forceinline__

typedef __attribute__((ext_vector_type(16))) _Float16 v16h;
typedef __attribute__((ext_vector_type(8)))  float    v8f;

namespace {
constexpr int B = 64, L = 4096, F = 64, D = 32, N = 4, K = 4, R = 2;
constexpr int T    = 64;          // time-chunk length
constexpr int HALO = 16;          // halo rows (only last K-1=3 actually feed the conv)
constexpr int ROWS = T + HALO;    // 80 rows staged per chunk
constexpr int C    = L / T;       // 64 chunks
constexpr int THREADS = 128;      // 4 waves (wave32)
constexpr int MT = ROWS / 16;     // 5 M-tiles of 16 rows
}

// ---- CDNA5 async memory->LDS copy (ASYNCcnt-tracked, no VGPR roundtrip) ---
// the LDS address comes from the VDST VGPR; generic pointers to LDS carry the LDS byte
// offset in their low 32 bits (aperture scheme), so truncation gives VDST.
DEV void async_copy_b128(void* lds_ptr, const void* gptr) {
  uint32_t lds_off = (uint32_t)(uintptr_t)lds_ptr;
  asm volatile("global_load_async_to_lds_b128 %0, %1, off"
               :: "v"(lds_off), "v"((uint64_t)(uintptr_t)gptr)
               : "memory");
}
DEV void wait_async0() {
  asm volatile("s_wait_asynccnt 0" ::: "memory");
}

// ---- WMMA fragment helpers (CDNA5 wave32 layouts, 16x16x32 f16) -----------
// A (16x32 f16): lane<16 -> row=lane, K = {0..7, 16..23}; lane>=16 -> row=lane-16,
// K = {8..15, 24..31}. Elements packed 2 halves / VGPR low-to-high.
DEV v16h load_a_frag_f16(const _Float16* base, int stride, int m0, int kbase, int lane) {
  int row = m0 + (lane & 15);
  int g0  = (lane & 16) ? 8 : 0;
  const _Float16* rp = base + row * stride + kbase + g0;
  v16h a;
  uint4* q = reinterpret_cast<uint4*>(&a);
  q[0] = *reinterpret_cast<const uint4*>(rp);        // k_local g0 .. g0+7
  q[1] = *reinterpret_cast<const uint4*>(rp + 16);   // k_local g0+16 .. g0+23
  return a;
}

// Same fragment built from f32 LDS (async-staged x), converting on the fly.
DEV v16h load_a_frag_f32(const float* base, int stride, int m0, int kbase, int lane) {
  int row = m0 + (lane & 15);
  int g0  = (lane & 16) ? 8 : 0;
  const float* rp = base + row * stride + kbase + g0;
  float4 f0 = *reinterpret_cast<const float4*>(rp);
  float4 f1 = *reinterpret_cast<const float4*>(rp + 4);
  float4 f2 = *reinterpret_cast<const float4*>(rp + 16);
  float4 f3 = *reinterpret_cast<const float4*>(rp + 20);
  v16h a;
  a[0]=(_Float16)f0.x; a[1]=(_Float16)f0.y; a[2]=(_Float16)f0.z; a[3]=(_Float16)f0.w;
  a[4]=(_Float16)f1.x; a[5]=(_Float16)f1.y; a[6]=(_Float16)f1.z; a[7]=(_Float16)f1.w;
  a[8]=(_Float16)f2.x; a[9]=(_Float16)f2.y; a[10]=(_Float16)f2.z; a[11]=(_Float16)f2.w;
  a[12]=(_Float16)f3.x; a[13]=(_Float16)f3.y; a[14]=(_Float16)f3.z; a[15]=(_Float16)f3.w;
  return a;
}

// B (32x16 f16): lanes 0-15 hold K={0..7,16..23} for column n=lane; lanes 16-31 hold
// K={8..15,24..31}.  We store B transposed as [n][k] so each lane reads two 8-K runs.
DEV v16h load_b_frag(const _Float16* baseT, int kdim, int n0, int kbase, int lane) {
  int col  = n0 + (lane & 15);
  int koff = kbase + ((lane & 16) ? 8 : 0);
  const _Float16* rp = baseT + col * kdim + koff;
  v16h bb;
  uint4* q = reinterpret_cast<uint4*>(&bb);
  q[0] = *reinterpret_cast<const uint4*>(rp);
  q[1] = *reinterpret_cast<const uint4*>(rp + 16);
  return bb;
}

// C/D (16x16 f32): VGPR j -> M = j + 8*(lane>=16), N = lane%16.
DEV void store_d_f16(_Float16* dst, int stride, int m0, int n0, int lane, v8f c) {
  int nn = n0 + (lane & 15);
  int mb = m0 + ((lane & 16) ? 8 : 0);
#pragma unroll
  for (int j = 0; j < 8; ++j) dst[(mb + j) * stride + nn] = (_Float16)c[j];
}
DEV void store_d_f32(float* dst, int stride, int m0, int n0, int lane, v8f c) {
  int nn = n0 + (lane & 15);
  int mb = m0 + ((lane & 16) ? 8 : 0);
#pragma unroll
  for (int j = 0; j < 8; ++j) dst[(mb + j) * stride + nn] = c[j];
}

DEV float sigmoidf_(float x) { return 1.f / (1.f + expf(-x)); }
DEV float softplusf_(float x) { return (x > 20.f) ? x : log1pf(expf(x)); }

// ===========================================================================
// Pass 1: one block per (chunk, batch). Async-stages x -> LDS, runs both GEMMs
// with v_wmma_f32_16x16x32_f16, conv+SiLU, projections, and emits the chunk's
// SSM summary (P = exp(A*sum dt), Q = chunk-local scan from h=0) to d_ws.
// ===========================================================================
__global__ __launch_bounds__(THREADS)
void ssm_pass1(const float* __restrict__ x,        const float* __restrict__ w_in,
               const float* __restrict__ b_in,     const float* __restrict__ w_inproj,
               const float* __restrict__ b_inproj, const float* __restrict__ conv_w,
               const float* __restrict__ conv_b,   const float* __restrict__ w_xproj,
               const float* __restrict__ w_dt,     const float* __restrict__ b_dt,
               const float* __restrict__ A_log,    const float* __restrict__ liquid_tau,
               float* __restrict__ wsP, float* __restrict__ wsQ, float* __restrict__ wsE)
{
  __shared__ __align__(16) float    sXf[ROWS * F];   // x chunk (f32, async-staged)
  __shared__ __align__(16) _Float16 sW1[D * F];      // w_in  as B^T: [d][f]
  __shared__ __align__(16) _Float16 sW2[D * D];      // w_inproj rows 0..31: [e][d]
  __shared__ __align__(16) _Float16 sH [ROWS * D];   // h (f16, A-operand of GEMM2)
  __shared__ __align__(16) float    sXi[ROWS * D];   // conv input
  __shared__ __align__(16) float    sXc[T * D];      // conv output (SiLU'd)
  __shared__ float sDt[T * D];
  __shared__ float sBs[T * N];
  __shared__ float sBin[D], sBinp[D], sConvB[D], sBdt[D], sTauSig[D];
  __shared__ float sConvW[D * K];
  __shared__ float sWxp[(R + 2 * N) * D];
  __shared__ float sWdt[D * R];

  const int tid  = threadIdx.x;
  const int lane = tid & 31;
  const int wave = tid >> 5;
  const int c  = blockIdx.x;
  const int b  = blockIdx.y;
  const int t0 = c * T;

  // ---- Phase 1a: kick off async DMA of x[t0-HALO .. t0+T) into LDS (f32).
  // Each b128 op moves one float4; negative (left-pad) rows are zero-filled.
  {
    const float* xb = x + (size_t)b * L * F;
    for (int v = tid; v < ROWS * (F / 4); v += THREADS) {
      int row = v >> 4;            // F/4 == 16 float4 per row
      int c4  = (v & 15) << 2;
      int gt  = t0 - HALO + row;
      if (gt >= 0) {
        async_copy_b128(&sXf[row * F + c4], xb + (size_t)gt * F + c4);
      } else {
        *reinterpret_cast<float4*>(&sXf[row * F + c4]) = make_float4(0.f, 0.f, 0.f, 0.f);
      }
    }
  }

  // ---- Phase 0: small weights -> LDS (overlaps with async x DMA)
  for (int i = tid; i < D * F; i += THREADS) sW1[i] = (_Float16)w_in[i];
  for (int i = tid; i < D * D; i += THREADS) sW2[i] = (_Float16)w_inproj[i];
  if (tid < D) {
    sBin[tid]    = b_in[tid];
    sBinp[tid]   = b_inproj[tid];
    sConvB[tid]  = conv_b[tid];
    sBdt[tid]    = b_dt[tid];
    sTauSig[tid] = sigmoidf_(liquid_tau[tid]);
  }
  for (int i = tid; i < D * K; i += THREADS)           sConvW[i] = conv_w[i];
  for (int i = tid; i < (R + 2 * N) * D; i += THREADS) sWxp[i]   = w_xproj[i];
  for (int i = tid; i < D * R; i += THREADS)           sWdt[i]   = w_dt[i];

  wait_async0();       // ASYNCcnt==0: all tensor bytes landed in LDS
  __syncthreads();     // cross-wave visibility

  // ---- Phase 2 (WMMA): h = x @ w_in^T + b_in   (M=ROWS, N=32, K=64)
  {
    v16h bfr[2][2];
#pragma unroll
    for (int kt = 0; kt < 2; ++kt)
#pragma unroll
      for (int nt = 0; nt < 2; ++nt)
        bfr[kt][nt] = load_b_frag(sW1, F, nt * 16, kt * 32, lane);

    for (int mt = wave; mt < MT; mt += 4) {      // wave-uniform: EXEC all ones
      int m0 = mt * 16;
      v16h a0 = load_a_frag_f32(sXf, F, m0, 0,  lane);
      v16h a1 = load_a_frag_f32(sXf, F, m0, 32, lane);
#pragma unroll
      for (int nt = 0; nt < 2; ++nt) {
        float bv = sBin[nt * 16 + (lane & 15)];
        v8f acc;
#pragma unroll
        for (int j = 0; j < 8; ++j) acc[j] = bv;
        acc = __builtin_amdgcn_wmma_f32_16x16x32_f16(false, a0, false, bfr[0][nt],
                                                     (short)0, acc, false, false);
        acc = __builtin_amdgcn_wmma_f32_16x16x32_f16(false, a1, false, bfr[1][nt],
                                                     (short)0, acc, false, false);
        store_d_f16(sH, D, m0, nt * 16, lane, acc);
      }
    }
  }
  __syncthreads();

  // ---- Phase 3 (WMMA): xi = h @ w_inproj[0:32]^T + b_inproj[0:32]  (K=32)
  {
    v16h bfr[2];
#pragma unroll
    for (int nt = 0; nt < 2; ++nt) bfr[nt] = load_b_frag(sW2, D, nt * 16, 0, lane);

    for (int mt = wave; mt < MT; mt += 4) {
      int m0 = mt * 16;
      v16h a = load_a_frag_f16(sH, D, m0, 0, lane);
#pragma unroll
      for (int nt = 0; nt < 2; ++nt) {
        float bv = sBinp[nt * 16 + (lane & 15)];
        v8f acc;
#pragma unroll
        for (int j = 0; j < 8; ++j) acc[j] = bv;
        acc = __builtin_amdgcn_wmma_f32_16x16x32_f16(false, a, false, bfr[nt],
                                                     (short)0, acc, false, false);
        store_d_f32(sXi, D, m0, nt * 16, lane, acc);
      }
    }
  }
  __syncthreads();

  // ---- Phase 4: causal depthwise conv (K=4, left zero-pad) + SiLU
  for (int idx = tid; idx < T * D; idx += THREADS) {
    int t = idx >> 5, d = idx & 31;
    int i = t + HALO;
    float acc = sConvB[d];
#pragma unroll
    for (int k = 0; k < K; ++k) {
      int j  = i - (K - 1) + k;
      int gt = t0 - HALO + j;
      float xv = (gt >= 0) ? sXi[j * D + d] : 0.f;
      acc = fmaf(sConvW[d * K + k], xv, acc);
    }
    sXc[t * D + d] = acc * sigmoidf_(acc);
  }
  __syncthreads();

  // ---- Phase 5: dbl projection -> dt (softplus * sigmoid(tau)), Bs, Cs
  if (tid < T) {
    const int t = tid;
    float xcv[D];
#pragma unroll
    for (int d = 0; d < D; ++d) xcv[d] = sXc[t * D + d];
    float dtp[R], Csv[N];
#pragma unroll
    for (int r = 0; r < R; ++r) {
      float s = 0.f;
#pragma unroll
      for (int d = 0; d < D; ++d) s = fmaf(xcv[d], sWxp[r * D + d], s);
      dtp[r] = s;
    }
#pragma unroll
    for (int n = 0; n < N; ++n) {
      float s = 0.f;
#pragma unroll
      for (int d = 0; d < D; ++d) s = fmaf(xcv[d], sWxp[(R + n) * D + d], s);
      sBs[t * N + n] = s;
    }
#pragma unroll
    for (int n = 0; n < N; ++n) {
      float s = 0.f;
#pragma unroll
      for (int d = 0; d < D; ++d) s = fmaf(xcv[d], sWxp[(R + N + n) * D + d], s);
      Csv[n] = s;
    }
#pragma unroll
    for (int d = 0; d < D; ++d) {
      float raw = fmaf(dtp[0], sWdt[d * R + 0], fmaf(dtp[1], sWdt[d * R + 1], sBdt[d]));
      sDt[t * D + d] = softplusf_(raw) * sTauSig[d];
    }
    if (t0 + t == L - 1) {           // export last-step xc and Cs for pass 2
#pragma unroll
      for (int d = 0; d < D; ++d) wsE[d] = xcv[d];
#pragma unroll
      for (int n = 0; n < N; ++n) wsE[64 + n] = Csv[n];
    }
  }
  __syncthreads();

  // ---- Phase 6: chunk-local linear scan summary per (d,n) thread
  {
    const int d = tid >> 2, n = tid & 3;
    const float Adn = -expf(A_log[d * N + n]);
    float h = 0.f, cum = 0.f;
    for (int t = 0; t < T; ++t) {
      float dtv = sDt[t * D + d];
      float dA  = expf(Adn * dtv);                       // independent of h: pipelines
      h   = fmaf(h, dA, dtv * sBs[t * N + n] * sXc[t * D + d]);
      cum += dtv;
    }
    size_t base = ((size_t)b * C + c) * THREADS + tid;
    wsP[base] = expf(Adn * cum);     // product of decays over the chunk
    wsQ[base] = h;                   // chunk-local contribution
  }

  // ---- Phase 7 (last chunk only): gate z at t=L-1 from h[L-1]
  if (c == C - 1 && tid < D) {
    int e = tid;
    float z = b_inproj[D + e];
#pragma unroll
    for (int d = 0; d < D; ++d)
      z = fmaf((float)sH[(ROWS - 1) * D + d], w_inproj[(D + e) * D + d], z);
    wsE[32 + e] = z;
  }
}

// ===========================================================================
// Pass 2: fold chunk summaries (64 per (b,d,n)), then the tiny head:
// y = (h.C + D_skip*xc) * silu(z);  y@w_out;  layernorm;  fc1+relu;  fc2.
// ===========================================================================
__global__ __launch_bounds__(THREADS)
void ssm_pass2(const float* __restrict__ wsP, const float* __restrict__ wsQ,
               const float* __restrict__ wsE,
               const float* __restrict__ D_skip, const float* __restrict__ w_out,
               const float* __restrict__ b_out,  const float* __restrict__ ln_g,
               const float* __restrict__ ln_b,   const float* __restrict__ w_fc1,
               const float* __restrict__ b_fc1,  const float* __restrict__ w_fc2,
               const float* __restrict__ b_fc2,  float* __restrict__ out)
{
  __shared__ float sRed[THREADS];
  __shared__ float sY[D];
  __shared__ float sO[D];
  __shared__ float sF[16];
  const int tid = threadIdx.x;
  const int b   = blockIdx.x;
  const int n   = tid & 3;

  float h = 0.f;
  for (int cc = 0; cc < C; ++cc) {
    size_t base = ((size_t)b * C + cc) * THREADS + tid;   // coalesced
    h = fmaf(h, wsP[base], wsQ[base]);
  }
  sRed[tid] = h * wsE[64 + n];       // h[d,n] * Cs_last[n]
  __syncthreads();

  if (tid < D) {
    int d = tid;
    float y = sRed[d * 4] + sRed[d * 4 + 1] + sRed[d * 4 + 2] + sRed[d * 4 + 3];
    y = fmaf(D_skip[d], wsE[d], y);
    float z = wsE[32 + d];
    y *= z * sigmoidf_(z);
    sY[d] = y;
  }
  __syncthreads();

  if (tid < D) {                      // lanes 0..31 of wave 0 (wave32): full wave
    int e = tid;
    float o = b_out[e];
#pragma unroll
    for (int d = 0; d < D; ++d) o = fmaf(sY[d], w_out[e * D + d], o);
    float s = o;
#pragma unroll
    for (int off = 16; off > 0; off >>= 1) s += __shfl_xor(s, off, 32);
    float mu = s * (1.f / 32.f);
    float dv = o - mu;
    float v2 = dv * dv;
#pragma unroll
    for (int off = 16; off > 0; off >>= 1) v2 += __shfl_xor(v2, off, 32);
    float var = v2 * (1.f / 32.f);
    sO[e] = dv * rsqrtf(var + 1e-5f) * ln_g[e] + ln_b[e];
  }
  __syncthreads();

  if (tid < 16) {
    float a = b_fc1[tid];
#pragma unroll
    for (int e = 0; e < D; ++e) a = fmaf(sO[e], w_fc1[tid * D + e], a);
    sF[tid] = fmaxf(a, 0.f);
  }
  __syncthreads();

  if (tid == 0) {
    float r = b_fc2[0];
#pragma unroll
    for (int j = 0; j < 16; ++j) r = fmaf(sF[j], w_fc2[j], r);
    out[b] = r;
  }
}

extern "C" void kernel_launch(void* const* d_in, const int* in_sizes, int n_in,
                              void* d_out, int out_size, void* d_ws, size_t ws_size,
                              hipStream_t stream) {
  (void)in_sizes; (void)n_in; (void)out_size; (void)ws_size;
  const float* x        = (const float*)d_in[0];
  const float* w_in     = (const float*)d_in[1];
  const float* b_in     = (const float*)d_in[2];
  const float* w_inproj = (const float*)d_in[3];
  const float* b_inproj = (const float*)d_in[4];
  const float* conv_w   = (const float*)d_in[5];
  const float* conv_b   = (const float*)d_in[6];
  const float* w_xproj  = (const float*)d_in[7];
  const float* w_dt     = (const float*)d_in[8];
  const float* b_dt     = (const float*)d_in[9];
  const float* A_log    = (const float*)d_in[10];
  const float* D_skip   = (const float*)d_in[11];
  const float* tau      = (const float*)d_in[12];
  const float* w_out    = (const float*)d_in[13];
  const float* b_out    = (const float*)d_in[14];
  const float* ln_g     = (const float*)d_in[15];
  const float* ln_b     = (const float*)d_in[16];
  const float* w_fc1    = (const float*)d_in[17];
  const float* b_fc1    = (const float*)d_in[18];
  const float* w_fc2    = (const float*)d_in[19];
  const float* b_fc2    = (const float*)d_in[20];

  float* wsP = (float*)d_ws;                              // [B][C][128]
  float* wsQ = wsP + (size_t)B * C * THREADS;             // [B][C][128]
  float* wsE = wsQ + (size_t)B * C * THREADS;             // [0:32)=xc_last, [32:64)=z_last, [64:68)=Cs_last

  dim3 g1(C, B);
  ssm_pass1<<<g1, THREADS, 0, stream>>>(x, w_in, b_in, w_inproj, b_inproj, conv_w,
                                        conv_b, w_xproj, w_dt, b_dt, A_log, tau,
                                        wsP, wsQ, wsE);
  ssm_pass2<<<B, THREADS, 0, stream>>>(wsP, wsQ, wsE, D_skip, w_out, b_out,
                                       ln_g, ln_b, w_fc1, b_fc1, w_fc2, b_fc2,
                                       (float*)d_out);
}
